// LinearAttention_11407433138823
// MI455X (gfx1250) — matrix operand order repaired and checked by reading, not compiled, not run on hardware
//
#include <hip/hip_runtime.h>

// ---------------------------------------------------------------------------
// LinearAttention for MI455X (gfx1250).
// All GEMMs via v_wmma_f32_16x16x32_bf16, fp32 accumulation.
// q/k/v/attn intermediates bf16 (~192MB working set ~= MI455X 192MB L2).
// - weights pre-converted to bf16 once; TDM (tensor_load_to_lds +
//   s_wait_tensorcnt) DMAs bf16 weight tiles to LDS -> frag reads are pure
//   ds_load_b128, zero per-iteration conversion VALU.
// - context GEMM and applyq A: LDS-free direct global_load_b128 fragments.
// - attn stored transposed (b,n,c) so the output projection B is also
//   direct-global (both operands k-contiguous).
// ---------------------------------------------------------------------------

typedef __attribute__((ext_vector_type(16))) __bf16          v16bf;
typedef __attribute__((ext_vector_type(16))) unsigned short  v16u;
typedef __attribute__((ext_vector_type(8)))  float           v8f;

#define NPIX 16384                           // H*W
#define BATCH 4
#define QELEMS (4ull * 512ull * 16384ull)    // elems per q/k/v buffer
#define CTXELEMS (16 * 128 * 128)            // context elements
#define WQ_ELEMS (1536 * 256)
#define WO_ELEMS (256 * 512)

#if defined(__gfx1250__) && __has_builtin(__builtin_amdgcn_tensor_load_to_lds) && \
    __has_builtin(__builtin_amdgcn_s_wait_tensorcnt)
#define HAVE_TDM 1
typedef __attribute__((ext_vector_type(4))) unsigned int tdm_g0_t;
typedef __attribute__((ext_vector_type(8))) int          tdm_g1_t;
typedef __attribute__((ext_vector_type(4))) int          tdm_g2_t;
#else
#define HAVE_TDM 0
#endif

__device__ __forceinline__ unsigned short f2bf(float f) {
  union { float f; unsigned u; } x; x.f = f;
  unsigned u = x.u + 0x7FFFu + ((x.u >> 16) & 1u);   // round-to-nearest-even
  return (unsigned short)(u >> 16);
}
__device__ __forceinline__ float bf2f(unsigned short h) {
  union { unsigned u; float f; } x; x.u = ((unsigned)h) << 16;
  return x.f;
}

#if HAVE_TDM
// 2D tile DMA: global (strided rows, d0 = contiguous dim) -> LDS, per
// CDNA5 ISA D# groups 0/1. Dims in data_size units.
// 6-arg builtin: (g0 x4u, g1 x8i, g2 x4i, g3 x4i, x8i, cpol)
__device__ __forceinline__ void tdm_load_2d(unsigned lds_addr, const void* gaddr,
                                            unsigned tile_d0, unsigned tile_d1,
                                            unsigned tensor_d0, unsigned tensor_d1,
                                            unsigned stride_d0, unsigned data_size_code)
{
  unsigned long long ga = (unsigned long long)gaddr;
  tdm_g0_t g0;
  g0[0] = 1u;                                              // count=1, user mode
  g0[1] = lds_addr;                                        // LDS byte address
  g0[2] = (unsigned)(ga & 0xFFFFFFFFu);                    // global_addr[31:0]
  g0[3] = (unsigned)((ga >> 32) & 0x01FFFFFFu) | (2u << 30); // addr[56:32] | type=2
  tdm_g1_t g1;
  g1[0] = (int)(data_size_code << 16);                     // wg_mask=0, data_size
  g1[1] = (int)((tensor_d0 & 0xFFFFu) << 16);              // tensor_dim0[15:0]
  g1[2] = (int)(((tensor_d0 >> 16) & 0xFFFFu) |
                ((tensor_d1 & 0xFFFFu) << 16));            // dim0 hi | dim1 lo
  g1[3] = (int)(((tensor_d1 >> 16) & 0xFFFFu) |
                ((tile_d0 & 0xFFFFu) << 16));              // dim1 hi | tile_dim0
  g1[4] = (int)(tile_d1 & 0xFFFFu);                        // tile_dim1, tile_dim2=0
  g1[5] = (int)stride_d0;                                  // dim0_stride[31:0]
  g1[6] = 0;
  g1[7] = 0;
  tdm_g2_t z4 = {0, 0, 0, 0};
  tdm_g1_t z8 = {0, 0, 0, 0, 0, 0, 0, 0};
  __builtin_amdgcn_tensor_load_to_lds(g0, g1, z4, z4, z8, 0);
}
#endif

// A fragment (16x32, MxK): lane m fixed; elems i -> k = 2*kbase + i (contiguous 32B)
__device__ __forceinline__ v16bf frag_a(const unsigned short* __restrict__ rowp,
                                        int kbase) {
  v16u u;
  #pragma unroll
  for (int i = 0; i < 16; ++i) u[i] = rowp[2 * kbase + i];
  return __builtin_bit_cast(v16bf, u);
}
// B fragment (32x16, KxN): lane n fixed; elems i -> k = grp*16 + i (contiguous 32B)
__device__ __forceinline__ v16bf frag_b(const unsigned short* __restrict__ colp,
                                        int grp) {
  v16u u;
  #pragma unroll
  for (int i = 0; i < 16; ++i) u[i] = colp[grp * 16 + i];
  return __builtin_bit_cast(v16bf, u);
}

#define WMMA_BF16(A, B, C) \
  __builtin_amdgcn_wmma_f32_16x16x32_bf16(false, (A), false, (B), (short)0, (C), false, false)

// ---------------------------------------------------------------------------
// One-time fp32 -> bf16 weight conversion (w_qkv and w_out).
// ---------------------------------------------------------------------------
__global__ __launch_bounds__(256)
void convert_weights_kernel(const float* __restrict__ wq,
                            const float* __restrict__ wo,
                            unsigned short* __restrict__ wq_bf,
                            unsigned short* __restrict__ wo_bf)
{
  int i = blockIdx.x * 256 + threadIdx.x;
  if (i < WQ_ELEMS)                    wq_bf[i] = f2bf(wq[i]);
  else if (i < WQ_ELEMS + WO_ELEMS)    wo_bf[i - WQ_ELEMS] = f2bf(wo[i - WQ_ELEMS]);
}

// ---------------------------------------------------------------------------
// QKV projection: qkv[b][m][n] = sum_k wq_bf[m][k] * x[b][k][n], rows split
// into q/k/v bf16. Block 64(M)x128(N); wave 32x32 (4 WMMAs/K-step).
// A: bf16 weight tile via TDM. B: x fp32 staged transposed [n][k] bf16.
// ---------------------------------------------------------------------------
__global__ __launch_bounds__(256)
void qkv_gemm_kernel(const unsigned short* __restrict__ Wbf,   // (1536,256) bf16
                     const float* __restrict__ x,              // (4,256,NPIX)
                     unsigned short* __restrict__ outQKV)
{
  const int n0    = blockIdx.x * 128;
  const int m0    = blockIdx.y * 64;
  const int b     = blockIdx.z;
  const int tid   = threadIdx.x;
  const int wave  = tid >> 5;
  const int lane  = tid & 31;
  const int wm    = wave >> 2;
  const int wn    = wave & 3;
  const int mlane = lane & 15;
  const int grp   = lane >> 4;
  const int kbase = grp * 8;
  const int KDIM  = 256;

  __shared__ unsigned short lAh[64 * 32];   // [m][k] bf16 (TDM target)
  __shared__ unsigned short lBt[128 * 32];  // [n][k] bf16 (transposed)

  v8f acc[2][2] = {};

  for (int k0 = 0; k0 < KDIM; k0 += 32) {
    // ---- stage A: 64x32 bf16 weight tile via Tensor Data Mover ----
#if HAVE_TDM
    if (wave == 0) {
      tdm_load_2d((unsigned)(unsigned long long)(void*)lAh,
                  &Wbf[(size_t)m0 * KDIM + k0],
                  /*tile_d0=*/32, /*tile_d1=*/64,
                  /*tensor_d0=*/KDIM, /*tensor_d1=*/1536,
                  /*stride_d0=*/KDIM, /*data_size=2B code*/1);
      __builtin_amdgcn_s_wait_tensorcnt(0);
    }
#else
    {
      int r = tid >> 2, c8 = (tid & 3) * 8;
      const unsigned short* src = &Wbf[(size_t)(m0 + r) * KDIM + k0 + c8];
      #pragma unroll
      for (int j = 0; j < 8; ++j) lAh[r * 32 + c8 + j] = src[j];
    }
#endif
    // ---- stage B transposed: 32(k) x 128(n) -> lBt[n][k], fp32->bf16 ----
    #pragma unroll
    for (int it = 0; it < 2; ++it) {
      int t = tid + it * 256;
      int r = t >> 4, c8 = (t & 15) * 8;
      size_t g = (size_t)b * KDIM * NPIX + (size_t)(k0 + r) * NPIX + n0 + c8;
      float tmp[8];
      #pragma unroll
      for (int j = 0; j < 8; ++j) tmp[j] = x[g + j];
      #pragma unroll
      for (int j = 0; j < 8; ++j) lBt[(c8 + j) * 32 + r] = f2bf(tmp[j]);
      if (k0 + 32 < KDIM) __builtin_prefetch(&x[g + 32 * NPIX], 0, 1);
    }
    __syncthreads();

    v16bf a0 = frag_a(&lAh[(wm * 32 + mlane) * 32], kbase);
    v16bf a1 = frag_a(&lAh[(wm * 32 + 16 + mlane) * 32], kbase);
    v16bf b0 = frag_b(&lBt[(wn * 32 + mlane) * 32], grp);
    v16bf b1 = frag_b(&lBt[(wn * 32 + 16 + mlane) * 32], grp);

    acc[0][0] = WMMA_BF16(a0, b0, acc[0][0]);
    acc[0][1] = WMMA_BF16(a0, b1, acc[0][1]);
    acc[1][0] = WMMA_BF16(a1, b0, acc[1][0]);
    acc[1][1] = WMMA_BF16(a1, b1, acc[1][1]);
    __syncthreads();
  }

  #pragma unroll
  for (int i = 0; i < 2; ++i)
    #pragma unroll
    for (int j = 0; j < 2; ++j)
      #pragma unroll
      for (int r = 0; r < 8; ++r) {
        int m = m0 + wm * 32 + i * 16 + grp * 8 + r;
        int n = n0 + wn * 32 + j * 16 + mlane;
        int part = m >> 9, mp = m & 511;          // 0=q 1=k 2=v ; mp = h*128+d
        outQKV[(size_t)part * QELEMS + (size_t)b * 512 * NPIX +
               (size_t)mp * NPIX + n] = f2bf(acc[i][j][r]);
      }
}

// ---------------------------------------------------------------------------
// Softmax over N per row of k (2048 rows x 16384), bf16 in/out, fp32 math.
// ---------------------------------------------------------------------------
__global__ __launch_bounds__(256)
void softmax_rows_kernel(unsigned short* kbuf)
{
  unsigned short* p = kbuf + (size_t)blockIdx.x * NPIX;
  __shared__ float red[256];
  const int tid = threadIdx.x;

  float m = -3.4e38f;
  for (int i = tid; i < NPIX; i += 256) m = fmaxf(m, bf2f(p[i]));
  red[tid] = m; __syncthreads();
  for (int s = 128; s > 0; s >>= 1) {
    if (tid < s) red[tid] = fmaxf(red[tid], red[tid + s]);
    __syncthreads();
  }
  m = red[0]; __syncthreads();

  float sum = 0.f;
  for (int i = tid; i < NPIX; i += 256) sum += __expf(bf2f(p[i]) - m);
  red[tid] = sum; __syncthreads();
  for (int s = 128; s > 0; s >>= 1) {
    if (tid < s) red[tid] += red[tid + s];
    __syncthreads();
  }
  float inv = 1.f / red[0];
  __syncthreads();

  for (int i = tid; i < NPIX; i += 256)
    p[i] = f2bf(__expf(bf2f(p[i]) - m) * inv);
}

__global__ __launch_bounds__(256)
void zero_f32_kernel(float* p, int n)
{
  int i = blockIdx.x * 256 + threadIdx.x;
  if (i < n) p[i] = 0.f;
}

// ---------------------------------------------------------------------------
// context[bh][d][e] += sum_{n chunk} ksm[bh][d][n] * v[bh][e][n]
// LDS-free: all fragments direct global_load_b128. Split-K over 32 chunks,
// fp32 atomic accumulate. Block: 8 waves = 8 d-tiles; wave 16(d)x64(e).
// ---------------------------------------------------------------------------
__global__ __launch_bounds__(256)
void context_wmma_kernel(const unsigned short* __restrict__ ksm,
                         const unsigned short* __restrict__ vbuf,
                         float* __restrict__ ctx)
{
  const int e0    = blockIdx.x * 64;
  const int chunk = blockIdx.y;
  const int bh    = blockIdx.z;
  const int wave  = threadIdx.x >> 5;
  const int lane  = threadIdx.x & 31;
  const int d0    = wave * 16;
  const int mlane = lane & 15;
  const int grp   = lane >> 4;
  const int kbase = grp * 8;

  const unsigned short* kb = ksm  + (size_t)bh * 128 * NPIX;
  const unsigned short* vb = vbuf + (size_t)bh * 128 * NPIX;
  const int nbeg = chunk * (NPIX / 32);

  v8f acc[4] = {};
  for (int n0 = nbeg; n0 < nbeg + NPIX / 32; n0 += 32) {
    v16bf a = frag_a(kb + (size_t)(d0 + mlane) * NPIX + n0, kbase);
    #pragma unroll
    for (int j = 0; j < 4; ++j) {
      v16bf bf = frag_b(vb + (size_t)(e0 + j * 16 + mlane) * NPIX + n0, grp);
      acc[j] = WMMA_BF16(a, bf, acc[j]);
    }
  }

  #pragma unroll
  for (int j = 0; j < 4; ++j)
    #pragma unroll
    for (int r = 0; r < 8; ++r)
      atomicAdd(&ctx[((size_t)bh * 128 + d0 + grp * 8 + r) * 128 +
                     e0 + j * 16 + mlane], acc[j][r]);
}

// ---------------------------------------------------------------------------
// One-time ctx fp32 [bh][d][e] -> bf16 transposed ctxT [bh][e][d]
// (makes applyq's A operand k-contiguous -> direct-global fragments).
// ---------------------------------------------------------------------------
__global__ __launch_bounds__(256)
void ctx_to_bf16_kernel(const float* __restrict__ ctxF,
                        unsigned short* __restrict__ ctxT)
{
  int i = blockIdx.x * 256 + threadIdx.x;       // (bh, e, d)
  if (i < CTXELEMS) {
    int bh = i >> 14, rem = i & 16383, e = rem >> 7, d = rem & 127;
    ctxT[i] = f2bf(ctxF[(bh << 14) + (d << 7) + e]);
  }
}

// ---------------------------------------------------------------------------
// attnT[b][n][c] = sum_d ctxT[bh][e][d] * q[bh][d][n]   (c = h*128+e)
// A: direct-global b128 fragments from ctxT (k-contiguous). B: q staged
// transposed [n][d]. Output transposed (b,n,c): contiguous per-lane stores.
// ---------------------------------------------------------------------------
__global__ __launch_bounds__(256)
void applyq_wmma_kernel(const unsigned short* __restrict__ ctxT,
                        const unsigned short* __restrict__ qbuf,
                        unsigned short* __restrict__ attnT)
{
  const int n0    = blockIdx.x * 128;
  const int e0    = blockIdx.y * 64;
  const int bh    = blockIdx.z;
  const int tid   = threadIdx.x;
  const int wave  = tid >> 5;
  const int lane  = tid & 31;
  const int wm    = wave >> 2;
  const int wn    = wave & 3;
  const int mlane = lane & 15;
  const int grp   = lane >> 4;
  const int kbase = grp * 8;

  __shared__ unsigned short lBt[128 * 32];  // [n][d]

  const unsigned short* qb = qbuf + (size_t)bh * 128 * NPIX;
  const unsigned short* ab = ctxT + (size_t)bh * 128 * 128;

  v8f acc[2][2] = {};

  for (int k0 = 0; k0 < 128; k0 += 32) {
    // stage B transposed: q[d][n] -> lBt[n][d]
    #pragma unroll
    for (int it = 0; it < 2; ++it) {
      int t = tid + it * 256;
      int r = t >> 4, c8 = (t & 15) * 8;
      const unsigned short* src = &qb[(size_t)(k0 + r) * NPIX + n0 + c8];
      unsigned short tmp[8];
      #pragma unroll
      for (int j = 0; j < 8; ++j) tmp[j] = src[j];
      #pragma unroll
      for (int j = 0; j < 8; ++j) lBt[(c8 + j) * 32 + r] = tmp[j];
    }
    __syncthreads();

    v16bf a0 = frag_a(ab + (size_t)(e0 + wm * 32 + mlane) * 128 + k0, kbase);
    v16bf a1 = frag_a(ab + (size_t)(e0 + wm * 32 + 16 + mlane) * 128 + k0, kbase);
    v16bf b0 = frag_b(&lBt[(wn * 32 + mlane) * 32], grp);
    v16bf b1 = frag_b(&lBt[(wn * 32 + 16 + mlane) * 32], grp);

    acc[0][0] = WMMA_BF16(a0, b0, acc[0][0]);
    acc[0][1] = WMMA_BF16(a0, b1, acc[0][1]);
    acc[1][0] = WMMA_BF16(a1, b0, acc[1][0]);
    acc[1][1] = WMMA_BF16(a1, b1, acc[1][1]);
    __syncthreads();
  }

  // transposed store: attnT[b][n][c], 8 contiguous u16 per (i,j)
  unsigned short* ob = attnT + (size_t)(bh >> 2) * NPIX * 512;
  const int cbase = (bh & 3) * 128 + e0 + wm * 32;
  #pragma unroll
  for (int i = 0; i < 2; ++i)
    #pragma unroll
    for (int j = 0; j < 2; ++j) {
      size_t nrow = (size_t)(n0 + wn * 32 + j * 16 + mlane) * 512;
      #pragma unroll
      for (int r = 0; r < 8; ++r)
        ob[nrow + cbase + i * 16 + grp * 8 + r] = f2bf(acc[i][j][r]);
    }
}

// ---------------------------------------------------------------------------
// Output projection: y[b][m][n] = sum_c wo_bf[m][c] * attnT[b][n][c] + bias[m]
// A: bf16 weight tile via TDM. B: direct-global fragments (k-contiguous).
// ---------------------------------------------------------------------------
__global__ __launch_bounds__(256)
void proj_gemm_kernel(const unsigned short* __restrict__ Wbf,   // (256,512) bf16
                      const unsigned short* __restrict__ attnT,
                      float* __restrict__ outF,
                      const float* __restrict__ bias)
{
  const int n0    = blockIdx.x * 128;
  const int m0    = blockIdx.y * 64;
  const int b     = blockIdx.z;
  const int tid   = threadIdx.x;
  const int wave  = tid >> 5;
  const int lane  = tid & 31;
  const int wm    = wave >> 2;
  const int wn    = wave & 3;
  const int mlane = lane & 15;
  const int grp   = lane >> 4;
  const int kbase = grp * 8;
  const int KDIM  = 512;

  __shared__ unsigned short lAh[64 * 32];   // [m][k] bf16 (TDM target)

  const unsigned short* bb = attnT + (size_t)b * NPIX * 512;

  v8f acc[2][2] = {};

  for (int k0 = 0; k0 < KDIM; k0 += 32) {
#if HAVE_TDM
    if (wave == 0) {
      tdm_load_2d((unsigned)(unsigned long long)(void*)lAh,
                  &Wbf[(size_t)m0 * KDIM + k0],
                  /*tile_d0=*/32, /*tile_d1=*/64,
                  /*tensor_d0=*/KDIM, /*tensor_d1=*/256,
                  /*stride_d0=*/KDIM, /*data_size=2B code*/1);
      __builtin_amdgcn_s_wait_tensorcnt(0);
    }
#else
    {
      int r = tid >> 2, c8 = (tid & 3) * 8;
      const unsigned short* src = &Wbf[(size_t)(m0 + r) * KDIM + k0 + c8];
      #pragma unroll
      for (int j = 0; j < 8; ++j) lAh[r * 32 + c8 + j] = src[j];
    }
#endif
    __syncthreads();

    v16bf a0 = frag_a(&lAh[(wm * 32 + mlane) * 32], kbase);
    v16bf a1 = frag_a(&lAh[(wm * 32 + 16 + mlane) * 32], kbase);
    // B fragments directly from global: attnT row n, k-contiguous
    v16bf b0 = frag_b(bb + (size_t)(n0 + wn * 32 + mlane) * 512 + k0, grp);
    v16bf b1 = frag_b(bb + (size_t)(n0 + wn * 32 + 16 + mlane) * 512 + k0, grp);

    acc[0][0] = WMMA_BF16(a0, b0, acc[0][0]);
    acc[0][1] = WMMA_BF16(a0, b1, acc[0][1]);
    acc[1][0] = WMMA_BF16(a1, b0, acc[1][0]);
    acc[1][1] = WMMA_BF16(a1, b1, acc[1][1]);
    __syncthreads();
  }

  #pragma unroll
  for (int i = 0; i < 2; ++i)
    #pragma unroll
    for (int j = 0; j < 2; ++j)
      #pragma unroll
      for (int r = 0; r < 8; ++r) {
        int m = m0 + wm * 32 + i * 16 + grp * 8 + r;
        int n = n0 + wn * 32 + j * 16 + mlane;
        outF[(size_t)b * 256 * NPIX + (size_t)m * NPIX + n] = acc[i][j][r] + bias[m];
      }
}

// ---------------------------------------------------------------------------
extern "C" void kernel_launch(void* const* d_in, const int* in_sizes, int n_in,
                              void* d_out, int out_size, void* d_ws, size_t ws_size,
                              hipStream_t stream)
{
  const float* x     = (const float*)d_in[0];   // (4,256,128,128)
  const float* w_qkv = (const float*)d_in[1];   // (1536,256)
  const float* w_out = (const float*)d_in[2];   // (256,512)
  const float* b_out = (const float*)d_in[3];   // (256,)
  float* y = (float*)d_out;                     // (4,256,128,128)

  unsigned short* q     = (unsigned short*)d_ws;        // 64 MB
  unsigned short* k     = q + QELEMS;                   // 64 MB
  unsigned short* v     = q + 2 * QELEMS;               // 64 MB
  float*          ctx   = (float*)(q + 3 * QELEMS);     // 1 MB fp32
  unsigned short* ctxT  = q + 3 * QELEMS + 2 * CTXELEMS;   // 512 KB bf16
  unsigned short* wq_bf = ctxT + CTXELEMS;              // 768 KB bf16
  unsigned short* wo_bf = wq_bf + WQ_ELEMS;             // 256 KB bf16
  unsigned short* attnT = v;   // v dead after context kernel; reuse region

  dim3 blk(256);

  // 0) one-time weight conversion to bf16
  convert_weights_kernel<<<dim3((WQ_ELEMS + WO_ELEMS) / 256), blk, 0, stream>>>(
      w_qkv, w_out, wq_bf, wo_bf);

  // 1) QKV projection: M=1536, K=256 -> q/k/v bf16
  qkv_gemm_kernel<<<dim3(NPIX / 128, 1536 / 64, BATCH), blk, 0, stream>>>(wq_bf, x, q);

  // 2) softmax over N on k
  softmax_rows_kernel<<<dim3(BATCH * 512), blk, 0, stream>>>(k);

  // 3) context = softmax(k) @ v^T  (split-K, fp32 atomic accumulate)
  zero_f32_kernel<<<dim3((CTXELEMS + 255) / 256), blk, 0, stream>>>(ctx, CTXELEMS);
  context_wmma_kernel<<<dim3(2, 32, 16), blk, 0, stream>>>(k, v, ctx);

  // 3b) ctx -> bf16 transposed (k-contiguous for applyq A)
  ctx_to_bf16_kernel<<<dim3((CTXELEMS + 255) / 256), blk, 0, stream>>>(ctx, ctxT);

  // 4) attnT = (ctx^T @ q), stored (b, n, c)
  applyq_wmma_kernel<<<dim3(NPIX / 128, 2, 16), blk, 0, stream>>>(ctxT, q, attnT);

  // 5) output projection: M=256, K=512, fp32 out + bias (B direct from global)
  proj_gemm_kernel<<<dim3(NPIX / 128, 256 / 64, BATCH), blk, 0, stream>>>(
      wo_bf, attnT, y, b_out);
}
